// Transformer_77790447665424
// MI455X (gfx1250) — matrix operand order repaired and checked by reading, not compiled, not run on hardware
//
#include <hip/hip_runtime.h>

// ---------------------------------------------------------------------------
// MI455X (gfx1250) transformer stack. Compute-bound (~110 GFLOP, ~100 MB
// traffic) -> everything runs on v_wmma_f32_16x16x32_bf16 (f32 accum).
// All global->LDS staging uses the CDNA5 async DMA path
// (global_load_async_to_lds_b128, ASYNCcnt) with LDS double buffering in both
// the GEMM and the flash-attention kernels. Weights packed column-major and V
// pre-transposed per head so every WMMA fragment load is k-contiguous
// (two ds_load_b128 per fragment, no scalar gathers).
// ---------------------------------------------------------------------------

typedef __bf16 bf;
typedef __attribute__((ext_vector_type(16))) __bf16 v16bf;
typedef __attribute__((ext_vector_type(8)))  float  v8f;

#define S_LEN   2048
#define D_MODEL 512
#define N_HEAD  8
#define D_HEAD  64
#define FF_DIM  2048
#define BATCH   2
#define M_ROWS  (BATCH * S_LEN)   // 4096

__device__ __forceinline__ bf f2bf(float f) {
  unsigned u = __float_as_uint(f);
  u += 0x7fffu + ((u >> 16) & 1u);           // round-to-nearest-even
  unsigned short s = (unsigned short)(u >> 16);
  bf r;
  __builtin_memcpy(&r, &s, sizeof(r));
  return r;
}

// Low 32 bits of a generic pointer to __shared__ data are the LDS byte offset
// (CDNA5 aperture rule).
__device__ __forceinline__ unsigned lds_off(const void* p) {
  return (unsigned)(uintptr_t)p;
}

// CDNA5 async DMA: 16 bytes per lane, global -> LDS, tracked by ASYNCcnt.
__device__ __forceinline__ void async_ld_b128(unsigned lds, const void* gptr) {
  asm volatile("global_load_async_to_lds_b128 %0, %1, off"
               :: "v"(lds), "v"((unsigned long long)(uintptr_t)gptr)
               : "memory");
}
__device__ __forceinline__ void wait_async0() {
  asm volatile("s_wait_asynccnt 0x0" ::: "memory");
}

// ---------------------------------------------------------------------------
// conversions / weight packing
// ---------------------------------------------------------------------------
__global__ void cvt_f32_bf16(const float* __restrict__ x, bf* __restrict__ y, int n) {
  int i = blockIdx.x * blockDim.x + threadIdx.x;
  if (i < n) y[i] = f2bf(x[i]);
}

// wq/wk/wv [H, D, DK] f32 -> column-major bf16 [N=H*DK][K=D]
__global__ void pack_qkv_t(const float* __restrict__ w, bf* __restrict__ out) {
  int i = blockIdx.x * blockDim.x + threadIdx.x;     // 512*512
  if (i < D_MODEL * D_MODEL) {
    int n = i >> 9;            // col  = h*64 + k
    int d = i & 511;           // row  (model dim, GEMM K)
    out[i] = f2bf(w[(((size_t)(n >> 6)) * D_MODEL + d) * D_HEAD + (n & 63)]);
  }
}

// generic [K,N] f32 row-major -> [N][K] bf16 (column-major pack)
__global__ void cvt_t_bf16(const float* __restrict__ in, bf* __restrict__ out,
                           int K, int N) {
  int i = blockIdx.x * blockDim.x + threadIdx.x;
  if (i < K * N) {
    int n = i / K, k = i - n * K;
    out[i] = f2bf(in[(size_t)k * N + n]);
  }
}

// V [B*S, H*DK] bf16 -> per-head d-major [B*H][64][S] so flash attention can
// async-DMA V tiles with k-contiguous fragment loads. LDS tile transpose,
// coalesced on both sides (pad 72 = 9*16B keeps b128 alignment, kills bank
// conflicts on the transposed stores).
__launch_bounds__(128)
__global__ void transpose_v(const bf* __restrict__ Vin, bf* __restrict__ Vt) {
  __shared__ alignas(16) bf tile[64][72];
  const int st  = blockIdx.x;               // s tile
  const int bh  = blockIdx.y;               // b*H + h
  const int b   = bh >> 3, h = bh & 7;
  const int tid = threadIdx.x;
#pragma unroll
  for (int i = 0; i < 4; ++i) {
    int idx = tid + i * 128;
    int r = idx >> 3, cp = idx & 7;         // r: s row, cp*8: d
    uint4 t = *(const uint4*)(Vin + ((size_t)(b * S_LEN + st * 64 + r)) * D_MODEL +
                              h * D_HEAD + cp * 8);
    const bf* tv = (const bf*)&t;
#pragma unroll
    for (int q = 0; q < 8; ++q) tile[cp * 8 + q][r] = tv[q];
  }
  __syncthreads();
#pragma unroll
  for (int i = 0; i < 4; ++i) {
    int idx = tid + i * 128;
    int r = idx >> 3, cp = idx & 7;         // r: d row, cp*8: s
    uint4 o;
    bf* ov = (bf*)&o;
#pragma unroll
    for (int q = 0; q < 8; ++q) ov[q] = tile[r][cp * 8 + q];
    *(uint4*)(Vt + ((size_t)(bh * 64 + r)) * S_LEN + st * 64 + cp * 8) = o;
  }
}

// ---------------------------------------------------------------------------
// WMMA GEMM: C[M,N] = A_bf16[M,K] @ Bt_bf16[N,K]^T  (+bias +scaled residual,
// optional relu), f32 and/or bf16 outputs. Block tile 64x128x32, 4 waves,
// each wave 32x64 (2x4 WMMA). Async global->LDS, double-buffered.
// ---------------------------------------------------------------------------
__launch_bounds__(128)
__global__ void gemm_bf16_wmma(const bf* __restrict__ A, int lda,
                               const bf* __restrict__ Bt, int ldb,
                               const float* __restrict__ bias,
                               const float* __restrict__ resid, float rscale,
                               int relu,
                               float* __restrict__ Cf, bf* __restrict__ Cb,
                               int N, int K) {
  __shared__ alignas(16) bf As[2][64][32];
  __shared__ alignas(16) bf Bs[2][128][32];  // [n][k] (weights pre-transposed)

  const int tid  = threadIdx.x;
  const int lane = tid & 31;
  const int wave = tid >> 5;
  const int wm   = wave >> 1;                // 0..1 (32 rows each)
  const int wn   = wave & 1;                 // 0..1 (64 cols each)
  const int bm   = blockIdx.y * 64;
  const int bn   = blockIdx.x * 128;
  const int half = lane >> 4;

  auto fill = [&](int bi, int k0) {
#pragma unroll
    for (int i = 0; i < 2; ++i) {            // A tile: 256 x 16B
      int idx = tid + i * 128;
      int r = idx >> 2, cp = idx & 3;
      async_ld_b128(lds_off(&As[bi][r][cp * 8]),
                    A + (size_t)(bm + r) * lda + k0 + cp * 8);
    }
#pragma unroll
    for (int i = 0; i < 4; ++i) {            // B tile: 512 x 16B
      int idx = tid + i * 128;
      int r = idx >> 2, cp = idx & 3;
      async_ld_b128(lds_off(&Bs[bi][r][cp * 8]),
                    Bt + (size_t)(bn + r) * ldb + k0 + cp * 8);
    }
  };

  v8f acc[2][4] = {};
  const int nPanels = K >> 5;

  fill(0, 0);
  wait_async0();
  __syncthreads();

  int buf = 0;
  for (int p = 0; p < nPanels; ++p) {
    if (p + 1 < nPanels) fill(buf ^ 1, (p + 1) * 32);  // DMA next panel

    // A frag: lane<16 K 0-15, lane>=16 K 16-31 (two b128 each)
    v16bf af[2], bfr[4];
#pragma unroll
    for (int tm = 0; tm < 2; ++tm) {
      int row = wm * 32 + tm * 16 + (lane & 15);
#pragma unroll
      for (int h = 0; h < 16; ++h)
        af[tm][h] = As[buf][row][h + (half << 4)];
    }
    // B frag from [n][k] tile: per-lane 16 contiguous bf16 (two b128)
#pragma unroll
    for (int tn = 0; tn < 4; ++tn) {
      int col = wn * 64 + tn * 16 + (lane & 15);
#pragma unroll
      for (int h = 0; h < 16; ++h)
        bfr[tn][h] = Bs[buf][col][h + (half << 4)];
    }
#pragma unroll
    for (int tm = 0; tm < 2; ++tm)
#pragma unroll
      for (int tn = 0; tn < 4; ++tn)
        acc[tm][tn] = __builtin_amdgcn_wmma_f32_16x16x32_bf16(
            false, af[tm], false, bfr[tn], (short)0, acc[tm][tn], false, false);

    wait_async0();       // next panel's DMA done
    __syncthreads();     // everyone finished reading current panel
    buf ^= 1;
  }

  // epilogue: C layout row = vgpr + 8*half, col = lane&15.
  // bias hoisted per column; residual loads batched so they overlap.
#pragma unroll
  for (int tm = 0; tm < 2; ++tm)
#pragma unroll
    for (int tn = 0; tn < 4; ++tn) {
      int rowb = bm + wm * 32 + tm * 16 + (half << 3);
      int col  = bn + wn * 64 + tn * 16 + (lane & 15);
      float bv = bias ? bias[col] : 0.0f;
      float rv[8];
      if (resid) {
#pragma unroll
        for (int r = 0; r < 8; ++r)
          rv[r] = resid[(size_t)(rowb + r) * N + col];
      }
#pragma unroll
      for (int r = 0; r < 8; ++r) {
        float v = acc[tm][tn][r] + bv;
        if (resid) v += rscale * rv[r];
        if (relu)  v = fmaxf(v, 0.0f);
        size_t off = (size_t)(rowb + r) * N + col;
        if (Cf) Cf[off] = v;
        if (Cb) Cb[off] = f2bf(v);
      }
    }
}

// ---------------------------------------------------------------------------
// Flash attention (online softmax), one block per (b, h, 64-row q tile).
// K and (pre-transposed) V tiles async-DMA'd with double buffering: one
// barrier per iteration, zero synchronous global loads in the hot loop.
// scores *= sqrt(DK) = 8 per reference.
// ---------------------------------------------------------------------------
__launch_bounds__(128)
__global__ void flash_attn_wmma(const bf* __restrict__ Q, const bf* __restrict__ Km,
                                const bf* __restrict__ Vt, bf* __restrict__ O,
                                int causal) {
  __shared__ alignas(16) bf Qs[64][64];
  __shared__ alignas(16) bf Ks[2][64][64];    // [s][d]
  __shared__ alignas(16) bf VsT[2][64][64];   // [d][s]
  __shared__ alignas(16) bf Ps[4][16][64];    // wave-private P staging

  const int tid  = threadIdx.x;
  const int lane = tid & 31;
  const int wave = tid >> 5;
  const int half = lane >> 4;
  const int qt = blockIdx.x, h = blockIdx.y, b = blockIdx.z;
  const int bh64 = (b * N_HEAD + h) * 64;

  const size_t qRow = (size_t)(b * S_LEN + qt * 64);
#pragma unroll
  for (int i = 0; i < 4; ++i) {
    int idx = tid + i * 128;
    int r = idx >> 3, cp = idx & 7;
    async_ld_b128(lds_off(&Qs[r][cp * 8]),
                  Q + (qRow + r) * D_MODEL + h * D_HEAD + cp * 8);
  }

  auto fillkv = [&](int bi, int j) {
    const size_t kvRow = (size_t)(b * S_LEN + j * 64);
#pragma unroll
    for (int i = 0; i < 4; ++i) {
      int idx = tid + i * 128;
      int r = idx >> 3, cp = idx & 7;
      async_ld_b128(lds_off(&Ks[bi][r][cp * 8]),
                    Km + (kvRow + r) * D_MODEL + h * D_HEAD + cp * 8);
      async_ld_b128(lds_off(&VsT[bi][r][cp * 8]),
                    Vt + ((size_t)(bh64 + r)) * S_LEN + j * 64 + cp * 8);
    }
  };

  v8f oacc[4] = {};
  float mrow[8], lrow[8];
#pragma unroll
  for (int r = 0; r < 8; ++r) { mrow[r] = -1e30f; lrow[r] = 0.0f; }

  const int jend = causal ? (qt + 1) : (S_LEN / 64);
  const int rbase = qt * 64 + wave * 16 + (half << 3);

  fillkv(0, 0);
  wait_async0();
  __syncthreads();

  int buf = 0;
  for (int j = 0; j < jend; ++j) {
    if (j + 1 < jend) fillkv(buf ^ 1, j + 1);          // DMA next tiles
    if (j + 2 < jend)                                   // L2 prefetch for j+2
      __builtin_prefetch(Km + ((size_t)(b * S_LEN + (j + 2) * 64) + (tid >> 1)) *
                                  D_MODEL + h * D_HEAD, 0, 1);

    // S = Q @ K^T   (B element (k,n) = Ks[n][k], k-contiguous per lane)
    v8f sacc[4] = {};
#pragma unroll
    for (int kk = 0; kk < 64; kk += 32) {
      v16bf af;
      int row = wave * 16 + (lane & 15);
#pragma unroll
      for (int hh = 0; hh < 16; ++hh)
        af[hh] = Qs[row][kk + hh + (half << 4)];
#pragma unroll
      for (int t = 0; t < 4; ++t) {
        v16bf bb;
        int col = t * 16 + (lane & 15);
#pragma unroll
        for (int hh = 0; hh < 16; ++hh)
          bb[hh] = Ks[buf][col][kk + hh + (half << 4)];
        sacc[t] = __builtin_amdgcn_wmma_f32_16x16x32_bf16(
            false, af, false, bb, (short)0, sacc[t], false, false);
      }
    }

    // scale + causal mask
#pragma unroll
    for (int t = 0; t < 4; ++t)
#pragma unroll
      for (int r = 0; r < 8; ++r) {
        float x = sacc[t][r] * 8.0f;
        if (causal) {
          int colg = j * 64 + t * 16 + (lane & 15);
          if (colg > rbase + r) x = -1e30f;
        }
        sacc[t][r] = x;
      }

    // online softmax: rows live across 16-lane halves -> shfl_xor width 16
#pragma unroll
    for (int r = 0; r < 8; ++r) {
      float mx = fmaxf(fmaxf(sacc[0][r], sacc[1][r]),
                       fmaxf(sacc[2][r], sacc[3][r]));
#pragma unroll
      for (int off = 1; off < 16; off <<= 1)
        mx = fmaxf(mx, __shfl_xor(mx, off, 16));
      float mn = fmaxf(mrow[r], mx);
      float alpha = __expf(mrow[r] - mn);
      mrow[r] = mn;
      float rs = 0.0f;
#pragma unroll
      for (int t = 0; t < 4; ++t) {
        float p = __expf(sacc[t][r] - mn);
        sacc[t][r] = p;
        rs += p;
      }
#pragma unroll
      for (int off = 1; off < 16; off <<= 1)
        rs += __shfl_xor(rs, off, 16);
      lrow[r] = lrow[r] * alpha + rs;
#pragma unroll
      for (int t = 0; t < 4; ++t) oacc[t][r] *= alpha;
    }

    // C-layout -> A-layout via wave-private LDS
#pragma unroll
    for (int t = 0; t < 4; ++t)
#pragma unroll
      for (int r = 0; r < 8; ++r)
        Ps[wave][r + (half << 3)][t * 16 + (lane & 15)] = f2bf(sacc[t][r]);
    asm volatile("s_wait_dscnt 0x0" ::: "memory");  // in-wave LDS RAW

    // O += P @ V   (B element (k,n) = VsT[n][k], k-contiguous per lane)
#pragma unroll
    for (int kk = 0; kk < 64; kk += 32) {
      v16bf pf;
      int row = lane & 15;
#pragma unroll
      for (int hh = 0; hh < 16; ++hh)
        pf[hh] = Ps[wave][row][kk + hh + (half << 4)];
#pragma unroll
      for (int t = 0; t < 4; ++t) {
        v16bf vf;
        int col = t * 16 + (lane & 15);
#pragma unroll
        for (int hh = 0; hh < 16; ++hh)
          vf[hh] = VsT[buf][col][kk + hh + (half << 4)];
        oacc[t] = __builtin_amdgcn_wmma_f32_16x16x32_bf16(
            false, pf, false, vf, (short)0, oacc[t], false, false);
      }
    }

    wait_async0();       // next tiles' DMA done
    __syncthreads();     // everyone finished reading current tiles
    buf ^= 1;
  }

  // normalize and write [B*S, H*DK] bf16
#pragma unroll
  for (int t = 0; t < 4; ++t)
#pragma unroll
    for (int r = 0; r < 8; ++r) {
      int row = qt * 64 + wave * 16 + (half << 3) + r;
      size_t off = ((size_t)(b * S_LEN + row)) * D_MODEL + h * D_HEAD +
                   t * 16 + (lane & 15);
      O[off] = f2bf(oacc[t][r] / lrow[r]);
    }
}

// ---------------------------------------------------------------------------
// LayerNorm over D=512 (no affine), one block per row, dual f32/bf16 output.
// ---------------------------------------------------------------------------
__launch_bounds__(256)
__global__ void layernorm_rows(const float* __restrict__ X,
                               float* __restrict__ Yf, bf* __restrict__ Yb) {
  __shared__ float s1[256], s2[256];
  const int row = blockIdx.x, tid = threadIdx.x;
  const float* xr = X + (size_t)row * D_MODEL;
  float a = xr[tid], c = xr[tid + 256];
  s1[tid] = a + c;
  s2[tid] = a * a + c * c;
  __syncthreads();
  for (int off = 128; off > 0; off >>= 1) {
    if (tid < off) { s1[tid] += s1[tid + off]; s2[tid] += s2[tid + off]; }
    __syncthreads();
  }
  float mean = s1[0] * (1.0f / D_MODEL);
  float var  = s2[0] * (1.0f / D_MODEL) - mean * mean;
  float rstd = rsqrtf(var + 1e-5f);
  float y0 = (a - mean) * rstd;
  float y1 = (c - mean) * rstd;
  size_t o0 = (size_t)row * D_MODEL + tid;
  if (Yf) { Yf[o0] = y0; Yf[o0 + 256] = y1; }
  if (Yb) { Yb[o0] = f2bf(y0); Yb[o0 + 256] = f2bf(y1); }
}

// ---------------------------------------------------------------------------
// host orchestration
// ---------------------------------------------------------------------------
extern "C" void kernel_launch(void* const* d_in, const int* in_sizes, int n_in,
                              void* d_out, int out_size, void* d_ws, size_t ws_size,
                              hipStream_t stream) {
  (void)in_sizes; (void)n_in; (void)out_size; (void)ws_size;

  const float* x       = (const float*)d_in[0];
  const float* wq_f[3] = {(const float*)d_in[1], (const float*)d_in[5], (const float*)d_in[9]};
  const float* wk_f[3] = {(const float*)d_in[2], (const float*)d_in[6], (const float*)d_in[10]};
  const float* wv_f[3] = {(const float*)d_in[3], (const float*)d_in[7], (const float*)d_in[11]};
  const float* wo_f[3] = {(const float*)d_in[4], (const float*)d_in[8], (const float*)d_in[12]};
  const float* w1_f[2] = {(const float*)d_in[13], (const float*)d_in[17]};
  const float* b1_f[2] = {(const float*)d_in[14], (const float*)d_in[18]};
  const float* w2_f[2] = {(const float*)d_in[15], (const float*)d_in[19]};
  const float* b2_f[2] = {(const float*)d_in[16], (const float*)d_in[20]};

  char* wp = (char*)d_ws;
  auto alloc = [&](size_t bytes) -> void* {
    void* p = (void*)wp;
    wp += (bytes + 255) & ~(size_t)255;
    return p;
  };

  const size_t MD = (size_t)M_ROWS * D_MODEL;
  const size_t MF = (size_t)M_ROWS * FF_DIM;
  const size_t DD = (size_t)D_MODEL * D_MODEL;
  const size_t DF = (size_t)D_MODEL * FF_DIM;

  bf* Xb = (bf*)alloc(MD * 2);
  bf *Wq[3], *Wk[3], *Wv[3], *Wo[3];
  for (int i = 0; i < 3; ++i) {
    Wq[i] = (bf*)alloc(DD * 2); Wk[i] = (bf*)alloc(DD * 2);
    Wv[i] = (bf*)alloc(DD * 2); Wo[i] = (bf*)alloc(DD * 2);
  }
  bf *W1[2], *W2[2];
  for (int i = 0; i < 2; ++i) { W1[i] = (bf*)alloc(DF * 2); W2[i] = (bf*)alloc(DF * 2); }
  bf* qb    = (bf*)alloc(MD * 2);
  bf* kb    = (bf*)alloc(MD * 2);
  bf* vb    = (bf*)alloc(MD * 2);
  bf* vT    = (bf*)alloc(MD * 2);     // [B*H][64][S] d-major
  bf* attnb = (bf*)alloc(MD * 2);
  bf* tb    = (bf*)alloc(MF * 2);
  bf* h1b = (bf*)alloc(MD * 2);
  bf* h2b = (bf*)alloc(MD * 2);
  bf* d1b = (bf*)alloc(MD * 2);
  bf* d2b = (bf*)alloc(MD * 2);
  float* tmpA = (float*)alloc(MD * 4);
  float* h1f  = (float*)alloc(MD * 4);
  float* d1f  = (float*)alloc(MD * 4);
  float* d2f  = (float*)alloc(MD * 4);

  // --- precision conversion / column-major weight packing ---
  cvt_f32_bf16<<<(int)((MD + 255) / 256), 256, 0, stream>>>(x, Xb, (int)MD);
  for (int i = 0; i < 3; ++i) {
    pack_qkv_t<<<(int)(DD / 256), 256, 0, stream>>>(wq_f[i], Wq[i]);
    pack_qkv_t<<<(int)(DD / 256), 256, 0, stream>>>(wk_f[i], Wk[i]);
    pack_qkv_t<<<(int)(DD / 256), 256, 0, stream>>>(wv_f[i], Wv[i]);
    cvt_t_bf16<<<(int)(DD / 256), 256, 0, stream>>>(wo_f[i], Wo[i], D_MODEL, D_MODEL);
  }
  for (int i = 0; i < 2; ++i) {
    cvt_t_bf16<<<(int)(DF / 256), 256, 0, stream>>>(w1_f[i], W1[i], D_MODEL, FF_DIM);
    cvt_t_bf16<<<(int)(DF / 256), 256, 0, stream>>>(w2_f[i], W2[i], FF_DIM, D_MODEL);
  }

  auto gemm = [&](const bf* A, int lda, const bf* Bt, const float* bias,
                  const float* resid, float rscale, int relu,
                  float* Cf, bf* Cb, int N, int K) {
    dim3 grid(N / 128, M_ROWS / 64);
    gemm_bf16_wmma<<<grid, 128, 0, stream>>>(A, lda, Bt, K, bias, resid, rscale,
                                             relu, Cf, Cb, N, K);
  };
  auto flash = [&](const bf* Qp, const bf* Kp, int causal) {
    transpose_v<<<dim3(S_LEN / 64, BATCH * N_HEAD), 128, 0, stream>>>(vb, vT);
    dim3 grid(S_LEN / 64, N_HEAD, BATCH);
    flash_attn_wmma<<<grid, 128, 0, stream>>>(Qp, Kp, vT, attnb, causal);
  };
  auto ln = [&](const float* in, float* of, bf* ob) {
    layernorm_rows<<<M_ROWS, 256, 0, stream>>>(in, of, ob);
  };

  // ---- encoder ----
  gemm(Xb, D_MODEL, Wq[0], nullptr, nullptr, 0.f, 0, nullptr, qb, D_MODEL, D_MODEL);
  gemm(Xb, D_MODEL, Wk[0], nullptr, nullptr, 0.f, 0, nullptr, kb, D_MODEL, D_MODEL);
  gemm(Xb, D_MODEL, Wv[0], nullptr, nullptr, 0.f, 0, nullptr, vb, D_MODEL, D_MODEL);
  flash(qb, kb, 0);
  gemm(attnb, D_MODEL, Wo[0], nullptr, x, 1.0f, 0, tmpA, nullptr, D_MODEL, D_MODEL);
  ln(tmpA, h1f, h1b);
  gemm(h1b, D_MODEL, W1[0], b1_f[0], nullptr, 0.f, 1, nullptr, tb, FF_DIM, D_MODEL);
  gemm(tb, FF_DIM, W2[0], b2_f[0], h1f, 1.0f, 0, tmpA, nullptr, D_MODEL, FF_DIM);
  ln(tmpA, nullptr, h2b);                       // enc_out (bf16 view suffices)

  // ---- decoder self-attention (causal, NO residual per reference) ----
  gemm(h2b, D_MODEL, Wq[1], nullptr, nullptr, 0.f, 0, nullptr, qb, D_MODEL, D_MODEL);
  gemm(h2b, D_MODEL, Wk[1], nullptr, nullptr, 0.f, 0, nullptr, kb, D_MODEL, D_MODEL);
  gemm(h2b, D_MODEL, Wv[1], nullptr, nullptr, 0.f, 0, nullptr, vb, D_MODEL, D_MODEL);
  flash(qb, kb, 1);
  gemm(attnb, D_MODEL, Wo[1], nullptr, nullptr, 0.f, 0, tmpA, nullptr, D_MODEL, D_MODEL);
  ln(tmpA, d1f, d1b);

  // ---- cross attention (q from decoder, k/v from enc_out) ----
  gemm(d1b, D_MODEL, Wq[2], nullptr, nullptr, 0.f, 0, nullptr, qb, D_MODEL, D_MODEL);
  gemm(h2b, D_MODEL, Wk[2], nullptr, nullptr, 0.f, 0, nullptr, kb, D_MODEL, D_MODEL);
  gemm(h2b, D_MODEL, Wv[2], nullptr, nullptr, 0.f, 0, nullptr, vb, D_MODEL, D_MODEL);
  flash(qb, kb, 0);
  gemm(attnb, D_MODEL, Wo[2], nullptr, d1f, 1.0f, 0, tmpA, nullptr, D_MODEL, D_MODEL);
  ln(tmpA, d2f, d2b);

  // ---- decoder FF: ln(d + ff(d)) == ln(2*d + relu(d@w1+b1)@w2 + b2) ----
  gemm(d2b, D_MODEL, W1[1], b1_f[1], nullptr, 0.f, 1, nullptr, tb, FF_DIM, D_MODEL);
  gemm(tb, FF_DIM, W2[1], b2_f[1], d2f, 2.0f, 0, tmpA, nullptr, D_MODEL, FF_DIM);
  ln(tmpA, (float*)d_out, nullptr);
}
